// FusionMambaBlock_62526133895656
// MI455X (gfx1250) — matrix operand order repaired and checked by reading, not compiled, not run on hardware
//
#include <hip/hip_runtime.h>
#include <hip/hip_fp16.h>

typedef __attribute__((ext_vector_type(16))) _Float16 v16h;
typedef __attribute__((ext_vector_type(8)))  float    v8f;

#define NB      4       // batch
#define HW      4096    // 64*64 tokens per image
#define BL      16384   // NB*HW tokens per stream
#define NR2     32768   // 2*BL tokens (both streams)
#define CM      96      // d_model
#define DI      192     // d_inner
#define NS      16      // d_state
#define DTR     6       // dt_rank

// ---------------- f32 -> f16 convert ----------------
__global__ void cvt_f16_kernel(const float* __restrict__ src, _Float16* __restrict__ dst, int n) {
    int i = blockIdx.x * blockDim.x + threadIdx.x;
    if (i < n) dst[i] = (_Float16)src[i];
}

// ---------------- Wfuse = conv2_w(96x96) @ out_proj_w(96x192) -> f16 (96,192) ----------------
__global__ void fuse_w_kernel(const float* __restrict__ c2w, const float* __restrict__ opw,
                              _Float16* __restrict__ wf) {
    int i = blockIdx.x * blockDim.x + threadIdx.x;
    if (i >= 96 * 192) return;
    int o = i / 192, d = i - o * 192;
    float acc = 0.f;
    for (int m = 0; m < 96; ++m) acc += c2w[o * 96 + m] * opw[m * 192 + d];
    wf[i] = (_Float16)acc;
}

// ---------------- NCHW f32 (two streams) -> token-major f16 (NR2, 96) ----------------
__global__ void im2tok_kernel(const float* __restrict__ x1, const float* __restrict__ x2,
                              _Float16* __restrict__ tok) {
    int i = blockIdx.x * blockDim.x + threadIdx.x;
    if (i >= NR2 * CM) return;
    int row = i / CM, c = i - row * CM;
    int s = row >> 14;          // stream
    int b = (row >> 12) & 3;    // batch
    int l = row & 4095;         // pixel
    const float* x = s ? x2 : x1;
    tok[i] = (_Float16)x[((size_t)(b * CM + c)) * HW + l];
}

// ---------------- WMMA GEMM: D(M,N) = A(M,K) * W(N,K)^T + bias ----------------
union FragU { v16h h; uint4 q[2]; };

__global__ void wmma_gemm_kernel(const _Float16* __restrict__ A, const _Float16* __restrict__ W,
                                 const float* __restrict__ bias, float* __restrict__ D,
                                 int M, int N, int K) {
    int wave = (blockIdx.x * blockDim.x + threadIdx.x) >> 5;
    int lane = threadIdx.x & 31;
    int ntile = N >> 4;
    int tm = wave / ntile;
    int tn = wave - tm * ntile;
    if (tm * 16 >= M) return;                    // wave-uniform
    int lh = lane >> 4;                          // half-wave select
    int ml = lane & 15;
    const _Float16* arow = A + (size_t)(tm * 16 + ml) * K;
    const _Float16* wrow = W + (size_t)(tn * 16 + ml) * K;
    v8f acc = {};
    for (int k0 = 0; k0 < K; k0 += 32) {
        FragU fa, fb;
        // A 16x32 layout: halves 0..7 <- K = k0+8*lh+j ; halves 8..15 <- K = k0+16+8*lh+j
        fa.q[0] = *(const uint4*)(arow + k0 + lh * 8);
        fa.q[1] = *(const uint4*)(arow + k0 + 16 + lh * 8);
        // B 32x16 layout: halves 0..7 <- K = k0+8*lh+j ; halves 8..15 <- K = k0+16+8*lh+j
        fb.q[0] = *(const uint4*)(wrow + k0 + lh * 8);
        fb.q[1] = *(const uint4*)(wrow + k0 + 16 + lh * 8);
        acc = __builtin_amdgcn_wmma_f32_16x16x32_f16(false, fa.h, false, fb.h,
                                                     (short)0, acc, false, false);
    }
    int ncol = tn * 16 + ml;
    float bv = bias ? bias[ncol] : 0.f;
#pragma unroll
    for (int r = 0; r < 8; ++r) {
        int mrow = tm * 16 + 8 * lh + r;
        D[(size_t)mrow * N + ncol] = acc[r] + bv;
    }
}

// ---------------- per-token channel LayerNorm (96), wave per token, f16 out ----------------
__global__ void ln_f16_kernel(const float* __restrict__ X, const float* __restrict__ g,
                              const float* __restrict__ bt, _Float16* __restrict__ Y, int T) {
    int wave = (blockIdx.x * blockDim.x + threadIdx.x) >> 5;
    int lane = threadIdx.x & 31;
    if (wave >= T) return;
    const float* row = X + (size_t)wave * CM;
    float x0 = row[lane], x1 = row[lane + 32], x2 = row[lane + 64];
    float s = x0 + x1 + x2, sq = x0 * x0 + x1 * x1 + x2 * x2;
    for (int o = 16; o >= 1; o >>= 1) { s += __shfl_xor(s, o, 32); sq += __shfl_xor(sq, o, 32); }
    float mean = s * (1.f / 96.f);
    float var  = sq * (1.f / 96.f) - mean * mean;
    float inv  = rsqrtf(var + 1e-5f);
    _Float16* yr = Y + (size_t)wave * CM;
    yr[lane]      = (_Float16)((x0 - mean) * inv * g[lane]      + bt[lane]);
    yr[lane + 32] = (_Float16)((x1 - mean) * inv * g[lane + 32] + bt[lane + 32]);
    yr[lane + 64] = (_Float16)((x2 - mean) * inv * g[lane + 64] + bt[lane + 64]);
}

// ---------------- depthwise causal conv(k=4) + bias + SiLU on u = XZ[:, :192] ----------------
__global__ void dwconv_silu_kernel(const float* __restrict__ XZ, const float* __restrict__ cw,
                                   const float* __restrict__ cb, float* __restrict__ U) {
    int i = blockIdx.x * blockDim.x + threadIdx.x;
    if (i >= NR2 * DI) return;
    int r = i / DI, d = i - r * DI;
    int l = r & 4095;                // position within its (stream,batch) sequence
    float acc = cb[d];
#pragma unroll
    for (int k = 0; k < 4; ++k) {
        int ll = l - 3 + k;
        if (ll >= 0) acc += cw[d * 4 + k] * XZ[(size_t)(r - 3 + k) * 384 + d];
    }
    float sg = 1.f / (1.f + __expf(-acc));
    U[i] = acc * sg;
}

// ---------------- per-token x_proj (192->38) + delta=softplus(dt@dtw^T+b) ----------------
__global__ void xproj_delta_kernel(const float* __restrict__ U, const float* __restrict__ xpw,
                                   const float* __restrict__ dtw, const float* __restrict__ dtb,
                                   float* __restrict__ Delta, float* __restrict__ Bm,
                                   float* __restrict__ Cm) {
    __shared__ float s_u[DI];
    __shared__ float s_x[DTR + 2 * NS];
    int t = blockIdx.x;
    int d = threadIdx.x;                  // 0..191
    s_u[d] = U[(size_t)t * DI + d];
    __syncthreads();
    if (d < DTR + 2 * NS) {
        float acc = 0.f;
        const float* wr = xpw + d * DI;
        for (int j = 0; j < DI; ++j) acc += wr[j] * s_u[j];
        s_x[d] = acc;
    }
    __syncthreads();
    float dt = dtb[d];
#pragma unroll
    for (int r = 0; r < DTR; ++r) dt += dtw[d * DTR + r] * s_x[r];
    float sp = fmaxf(dt, 0.f) + log1pf(__expf(-fabsf(dt)));   // stable softplus
    Delta[(size_t)t * DI + d] = sp;
    if (d < NS) { Bm[t * NS + d] = s_x[DTR + d]; Cm[t * NS + d] = s_x[DTR + NS + d]; }
}

// ---------------- selective scan: h = exp(dl*A)h + dl*B*u ; y = <h,C> ; gate ----------------
// grid = 96 blocks (8 seqs * 12 channel-groups), 256 threads; each wave: 2 channels x 16 states
__global__ void scan_kernel(const float* __restrict__ Delta, const float* __restrict__ U,
                            const float* __restrict__ Bm, const float* __restrict__ Cm,
                            const float* __restrict__ XZ, const float* __restrict__ A_log,
                            const float* __restrict__ Dp, float* __restrict__ Y) {
    int sb = blockIdx.x / 12;                 // 0..7 = stream*4 + batch
    int cb = (blockIdx.x % 12) * 16;          // channel group base
    int wave = threadIdx.x >> 5;
    int lane = threadIdx.x & 31;
    int d = cb + wave * 2 + (lane >> 4);      // d_inner channel
    int n = lane & 15;                        // state index
    float A   = -__expf(A_log[d * NS + n]);
    float dpv = Dp[d];
    float h = 0.f;
    size_t r0 = (size_t)sb * HW;
    for (int l = 0; l < HW; ++l) {
        size_t r = r0 + l;
        float dl = Delta[r * DI + d];
        float uv = U[r * DI + d];
        float bn = Bm[r * NS + n];
        float cn = Cm[r * NS + n];
        h = __expf(dl * A) * h + dl * bn * uv;
        float p = h * cn;
        p += __shfl_xor(p, 8, 32);            // reduce over 16-lane state group
        p += __shfl_xor(p, 4, 32);
        p += __shfl_xor(p, 2, 32);
        p += __shfl_xor(p, 1, 32);
        if (n == 0) {
            float yv = p + uv * dpv;
            float z  = XZ[r * 384 + DI + d];
            float sg = 1.f / (1.f + __expf(-z));
            Y[r * DI + d] = yv * (z * sg);
        }
    }
}

// ---------------- sum both streams -> f16 (BL,192) ----------------
__global__ void combine_kernel(const float* __restrict__ Y, _Float16* __restrict__ Yc) {
    int i = blockIdx.x * blockDim.x + threadIdx.x;
    if (i >= BL * DI) return;
    Yc[i] = (_Float16)(Y[i] + Y[(size_t)BL * DI + i]);
}

// ---------------- final LN over channels + transpose back to NCHW f32 ----------------
__global__ void ln_out_kernel(const float* __restrict__ X, const float* __restrict__ g,
                              const float* __restrict__ bt, float* __restrict__ Out) {
    int wave = (blockIdx.x * blockDim.x + threadIdx.x) >> 5;
    int lane = threadIdx.x & 31;
    if (wave >= BL) return;
    const float* row = X + (size_t)wave * CM;
    float x0 = row[lane], x1 = row[lane + 32], x2 = row[lane + 64];
    float s = x0 + x1 + x2, sq = x0 * x0 + x1 * x1 + x2 * x2;
    for (int o = 16; o >= 1; o >>= 1) { s += __shfl_xor(s, o, 32); sq += __shfl_xor(sq, o, 32); }
    float mean = s * (1.f / 96.f);
    float var  = sq * (1.f / 96.f) - mean * mean;
    float inv  = rsqrtf(var + 1e-5f);
    int b = wave >> 12, l = wave & 4095;
    float* ob = Out + (size_t)b * CM * HW;
    ob[(size_t)lane * HW + l]        = (x0 - mean) * inv * g[lane]      + bt[lane];
    ob[(size_t)(lane + 32) * HW + l] = (x1 - mean) * inv * g[lane + 32] + bt[lane + 32];
    ob[(size_t)(lane + 64) * HW + l] = (x2 - mean) * inv * g[lane + 64] + bt[lane + 64];
}

// =====================================================================
extern "C" void kernel_launch(void* const* d_in, const int* in_sizes, int n_in,
                              void* d_out, int out_size, void* d_ws, size_t ws_size,
                              hipStream_t stream) {
    const float* x1      = (const float*)d_in[0];
    const float* x2      = (const float*)d_in[1];
    const float* conv1_w = (const float*)d_in[2];
    const float* conv1_b = (const float*)d_in[3];
    const float* norm1_w = (const float*)d_in[4];
    const float* norm1_b = (const float*)d_in[5];
    const float* in_proj_w = (const float*)d_in[6];
    const float* conv1d_w  = (const float*)d_in[7];
    const float* conv1d_b  = (const float*)d_in[8];
    const float* x_proj_w  = (const float*)d_in[9];
    const float* dt_proj_w = (const float*)d_in[10];
    const float* dt_proj_b = (const float*)d_in[11];
    const float* A_log     = (const float*)d_in[12];
    const float* Dp        = (const float*)d_in[13];
    const float* out_proj_w = (const float*)d_in[14];
    const float* conv2_w   = (const float*)d_in[15];
    const float* conv2_b   = (const float*)d_in[16];
    const float* norm2_w   = (const float*)d_in[17];
    const float* norm2_b   = (const float*)d_in[18];
    float* out = (float*)d_out;

    // ---- workspace carve-up (cursor, 256B aligned) ----
    char* ws = (char*)d_ws;
    size_t cur = 0;
    auto alloc = [&](size_t bytes) { void* p = ws + cur; cur += (bytes + 255) & ~(size_t)255; return p; };

    _Float16* tok_h  = (_Float16*)alloc((size_t)NR2 * CM * 2);   // tokens f16
    _Float16* wA_h   = (_Float16*)alloc(96 * 96 * 2);            // conv1_w f16
    _Float16* wIn_h  = (_Float16*)alloc(384 * 96 * 2);           // in_proj_w f16
    _Float16* wF_h   = (_Float16*)alloc(96 * 192 * 2);           // conv2 o out_proj f16
    float*    O1     = (float*)alloc((size_t)NR2 * CM * 4);      // conv1 out (also reused for final GEMM out)
    _Float16* ln_h   = (_Float16*)alloc((size_t)NR2 * CM * 2);   // LN1 out f16
    float*    XZ     = (float*)alloc((size_t)NR2 * 384 * 4);     // in_proj out [u|z]
    float*    Ubuf   = (float*)alloc((size_t)NR2 * DI * 4);      // conv1d+silu
    float*    Dl     = (float*)alloc((size_t)NR2 * DI * 4);      // delta
    float*    BmB    = (float*)alloc((size_t)NR2 * NS * 4);
    float*    CmB    = (float*)alloc((size_t)NR2 * NS * 4);
    float*    Ybuf   = (float*)alloc((size_t)NR2 * DI * 4);      // scan out per stream
    _Float16* Yc_h   = (_Float16*)alloc((size_t)BL * DI * 2);    // stream sum f16
    float*    OUT2   = O1;                                       // reuse O1 region for (BL,96) final GEMM out
    (void)ws_size; (void)n_in; (void)in_sizes; (void)out_size;

    // ---- weight conversions + fusion ----
    cvt_f16_kernel<<<(96 * 96 + 255) / 256, 256, 0, stream>>>(conv1_w, wA_h, 96 * 96);
    cvt_f16_kernel<<<(384 * 96 + 255) / 256, 256, 0, stream>>>(in_proj_w, wIn_h, 384 * 96);
    fuse_w_kernel<<<(96 * 192 + 255) / 256, 256, 0, stream>>>(conv2_w, out_proj_w, wF_h);

    // ---- tokens ----
    im2tok_kernel<<<((size_t)NR2 * CM + 255) / 256, 256, 0, stream>>>(x1, x2, tok_h);

    // ---- GEMM1: conv1 1x1 (+bias) on both streams: (32768,96) ----
    {
        int waves = (NR2 / 16) * (CM / 16);
        wmma_gemm_kernel<<<(waves * 32 + 255) / 256, 256, 0, stream>>>(tok_h, wA_h, conv1_b, O1, NR2, CM, CM);
    }
    // ---- LN1 -> f16 ----
    ln_f16_kernel<<<(NR2 * 32 + 255) / 256, 256, 0, stream>>>(O1, norm1_w, norm1_b, ln_h, NR2);

    // ---- GEMM2: in_proj: (32768,384) ----
    {
        int waves = (NR2 / 16) * (384 / 16);
        wmma_gemm_kernel<<<(waves * 32 + 255) / 256, 256, 0, stream>>>(ln_h, wIn_h, nullptr, XZ, NR2, 384, CM);
    }
    // ---- depthwise conv + SiLU ----
    dwconv_silu_kernel<<<((size_t)NR2 * DI + 255) / 256, 256, 0, stream>>>(XZ, conv1d_w, conv1d_b, Ubuf);
    // ---- x_proj + delta (block per token) ----
    xproj_delta_kernel<<<NR2, DI, 0, stream>>>(Ubuf, x_proj_w, dt_proj_w, dt_proj_b, Dl, BmB, CmB);
    // ---- selective scan + gate ----
    scan_kernel<<<96, 256, 0, stream>>>(Dl, Ubuf, BmB, CmB, XZ, A_log, Dp, Ybuf);
    // ---- sum streams -> f16 ----
    combine_kernel<<<((size_t)BL * DI + 255) / 256, 256, 0, stream>>>(Ybuf, Yc_h);

    // ---- GEMM3: fused (conv2 o out_proj) + conv2_b: (16384,96) ----
    {
        int waves = (BL / 16) * (CM / 16);
        wmma_gemm_kernel<<<(waves * 32 + 255) / 256, 256, 0, stream>>>(Yc_h, wF_h, conv2_b, OUT2, BL, CM, DI);
    }
    // ---- LN2 + transpose to NCHW ----
    ln_out_kernel<<<(BL * 32 + 255) / 256, 256, 0, stream>>>(OUT2, norm2_w, norm2_b, out);
}
